// CAModel_2121713844629
// MI455X (gfx1250) — matrix operand order repaired and checked by reading, not compiled, not run on hardware
//
#include <hip/hip_runtime.h>
#include <stdint.h>

// ---------------------------------------------------------------------------
// NCA step for MI455X (gfx1250, wave32).
//   Kernel 1: per 16x16 spatial tile: halo load -> pre_life + depthwise 3x3
//             perceive (fp32 VALU) -> 64->128->16 MLP via v_wmma_f32_16x16x32_bf16
//             -> x_new written straight to d_out; alpha plane + pre mask to ws.
//   Kernel 2: 3x3 stencil on updated alpha (from ws) -> post_life; zeroes the
//             16 channels of dead pixels in d_out (live pixels already final).
// ---------------------------------------------------------------------------

typedef __attribute__((ext_vector_type(16))) __bf16 v16bf;
typedef __attribute__((ext_vector_type(8)))  float  v8f;

#define HDIM 128
#define WDIM 128
#define CCH  16
#define NFEAT 4
#define HID  128
#define FANIN 64   // CCH * NFEAT

__device__ __forceinline__ unsigned short f2bf(float f) {
  union { float f; uint32_t u; } v; v.f = f;
  uint32_t u = v.u;
  uint32_t r = u + 0x7FFFu + ((u >> 16) & 1u);   // round-to-nearest-even
  return (unsigned short)(r >> 16);
}

union FragU { v16bf bf; uint32_t u[8]; };

// A-operand (16x32 bf16). rowbase points at this lane's matrix row (row = lane&15
// within the M-tile). ISA layout: lanes 0-15 hold K=k0+[0..7] and k0+[16..23],
// lanes 16-31 hold K=k0+[8..15] and k0+[24..31].
__device__ __forceinline__ v16bf load_fragA(const unsigned short* rowbase, int k0, int lane) {
  int off = k0 + ((lane & 16) ? 8 : 0);
  FragU f;
  const uint32_t* p = (const uint32_t*)(rowbase + off);
#pragma unroll
  for (int i = 0; i < 4; ++i) f.u[i] = p[i];
  const uint32_t* q = (const uint32_t*)(rowbase + off + 16);
#pragma unroll
  for (int i = 0; i < 4; ++i) f.u[4 + i] = q[i];
  return f.bf;
}

// B-operand (32x16 bf16) given B^T stored row-major (rowbase = row n of B^T,
// i.e. column n of B). ISA layout: lanes 0-15 hold K=k0+[0..7] and k0+[16..23],
// lanes 16-31 hold K=k0+[8..15] and k0+[24..31].
__device__ __forceinline__ v16bf load_fragB(const unsigned short* rowbase, int k0, int lane) {
  int off = k0 + ((lane & 16) ? 8 : 0);
  FragU f;
  const uint32_t* p = (const uint32_t*)(rowbase + off);
#pragma unroll
  for (int i = 0; i < 4; ++i) f.u[i] = p[i];
  const uint32_t* q = (const uint32_t*)(rowbase + off + 16);
#pragma unroll
  for (int i = 0; i < 4; ++i) f.u[4 + i] = q[i];
  return f.bf;
}

__global__ __launch_bounds__(256) void nca_step_kernel(
    const float* __restrict__ x, const float* __restrict__ rmask,
    const float* __restrict__ filters, const float* __restrict__ W1,
    const float* __restrict__ b1, const float* __restrict__ W2,
    const float* __restrict__ b2, float* __restrict__ out,
    float* __restrict__ aws, unsigned char* __restrict__ pre) {
  __shared__ float          xS[CCH][18][18];      // 20736 B  halo tile, fp32
  __shared__ unsigned short w1s[HID * FANIN];     // 16384 B  W1 (B^T row-major) bf16
  __shared__ unsigned short w2s[CCH * HID];       //  4096 B  W2 (B^T row-major) bf16
  __shared__ float          b1s[HID];
  __shared__ float          b2s[CCH];
  __shared__ float          filtS[NFEAT][3][3];
  __shared__ unsigned short yS[8][32][FANIN];     // 32768 B  per-wave perceive features
  __shared__ unsigned short hS[8][16][HID];       // 32768 B  per-wave hidden (bf16)
  __shared__ float          dxS[8][32][CCH];      // 16384 B  per-wave dx, C-layout unscrambled

  const int tid  = threadIdx.x;
  const int b    = blockIdx.x >> 6;
  const int tile = blockIdx.x & 63;
  const int th0  = (tile >> 3) << 4;
  const int tw0  = (tile & 7) << 4;

  // ---- load halo tile + weights into LDS ----
  for (int i = tid; i < CCH * 18 * 18; i += 256) {
    int c = i / 324, rem = i % 324;
    int yy = rem / 18, xx = rem % 18;
    int gh = (th0 + yy - 1) & (HDIM - 1);
    int gw = (tw0 + xx - 1) & (WDIM - 1);
    xS[c][yy][xx] = x[((b * CCH + c) * HDIM + gh) * WDIM + gw];
  }
  for (int i = tid; i < HID * FANIN; i += 256) w1s[i] = f2bf(W1[i]);
  for (int i = tid; i < CCH * HID;  i += 256) w2s[i] = f2bf(W2[i]);
  if (tid < HID) b1s[tid] = b1[tid];
  if (tid < CCH) b2s[tid] = b2[tid];
  if (tid < NFEAT * 9) ((float*)filtS)[tid] = filters[tid];
  __syncthreads();

  const int lane = tid & 31;
  const int wid  = tid >> 5;
  const int py   = tid >> 4;   // tile row of this thread's pixel
  const int px   = tid & 15;   // tile col
  const int gh   = th0 + py, gw = tw0 + px;

  // ---- phase 1: pre_life + perceive (one pixel per thread) ----
  {
    float amax = -3.4e38f, asum = 0.f;
#pragma unroll
    for (int j = 0; j < 3; ++j)
#pragma unroll
      for (int i = 0; i < 3; ++i) {
        float v = xS[3][py + j][px + i];
        amax = fmaxf(amax, v); asum += v;
      }
    int prelife = (amax > 0.1f) && (asum * (1.f / 9.f) < 0.2f);
    pre[(b * HDIM + gh) * WDIM + gw] = (unsigned char)prelife;

#pragma unroll
    for (int c = 0; c < CCH; ++c) {
      float s0 = 0.f, s1 = 0.f, s2 = 0.f, s3 = 0.f;
#pragma unroll
      for (int j = 0; j < 3; ++j)
#pragma unroll
        for (int i = 0; i < 3; ++i) {
          float v = xS[c][py + j][px + i];
          s0 += v * filtS[0][j][i];
          s1 += v * filtS[1][j][i];
          s2 += v * filtS[2][j][i];
          s3 += v * filtS[3][j][i];
        }
      uint32_t* dst = (uint32_t*)&yS[wid][lane][4 * c];
      dst[0] = (uint32_t)f2bf(s0) | ((uint32_t)f2bf(s1) << 16);
      dst[1] = (uint32_t)f2bf(s2) | ((uint32_t)f2bf(s3) << 16);
    }
  }
  const float um = (rmask[(b * HDIM + gh) * WDIM + gw] <= 0.5f) ? 1.f : 0.f;

  // ---- phase 2: MLP via WMMA, two 16-pixel M-tiles per wave ----
  const v8f vzero = {0.f, 0.f, 0.f, 0.f, 0.f, 0.f, 0.f, 0.f};
  const int mhi = (lane & 16) ? 8 : 0;   // C/D layout: lanes16-31 hold rows M+8
  const int nlo = lane & 15;             // C/D layout: N = lane&15

#pragma unroll 1
  for (int t = 0; t < 2; ++t) {
    // layer 1: h[16x128] = y[16x64] * W1^T[64x128]
    v8f acc[8];
#pragma unroll
    for (int n = 0; n < 8; ++n) acc[n] = vzero;
    const unsigned short* yrow = &yS[wid][t * 16 + nlo][0];
#pragma unroll
    for (int k = 0; k < 2; ++k) {
      v16bf a = load_fragA(yrow, k * 32, lane);
#pragma unroll
      for (int n = 0; n < 8; ++n) {
        v16bf bb = load_fragB(&w1s[(n * 16 + nlo) * FANIN], k * 32, lane);
        acc[n] = __builtin_amdgcn_wmma_f32_16x16x32_bf16(
            false, a, false, bb, (short)0, acc[n], false, false);
      }
    }
    // bias + leaky ReLU, restage hidden as bf16 A operand
#pragma unroll
    for (int n = 0; n < 8; ++n) {
      float bv = b1s[n * 16 + nlo];
#pragma unroll
      for (int r = 0; r < 8; ++r) {
        float v = acc[n][r] + bv;
        v = (v >= 0.f) ? v : 0.01f * v;
        hS[wid][r + mhi][n * 16 + nlo] = f2bf(v);
      }
    }
    // layer 2: dx[16x16] = h[16x128] * W2^T[128x16]
    v8f acc2 = vzero;
    const unsigned short* hrow = &hS[wid][nlo][0];
#pragma unroll
    for (int k = 0; k < 4; ++k) {
      v16bf a  = load_fragA(hrow, k * 32, lane);
      v16bf bb = load_fragB(&w2s[nlo * HID], k * 32, lane);
      acc2 = __builtin_amdgcn_wmma_f32_16x16x32_bf16(
          false, a, false, bb, (short)0, acc2, false, false);
    }
    float b2v = b2s[nlo];
#pragma unroll
    for (int r = 0; r < 8; ++r)
      dxS[wid][t * 16 + r + mhi][nlo] = acc2[r] + b2v;
  }

  // ---- phase 3: masked update; x_new -> d_out, alpha_new -> workspace ----
  const float* dxp = &dxS[wid][lane][0];
#pragma unroll
  for (int c = 0; c < CCH; ++c) {
    float xn = xS[c][py + 1][px + 1] + dxp[c] * um;   // STEP_SIZE == 1
    out[((b * CCH + c) * HDIM + gh) * WDIM + gw] = xn;
    if (c == 3) aws[(b * HDIM + gh) * WDIM + gw] = xn;
  }
}

__global__ __launch_bounds__(256) void nca_final_kernel(
    const float* __restrict__ aws, const unsigned char* __restrict__ pre,
    float* __restrict__ out) {
  __shared__ float aS[18][18];
  const int tid  = threadIdx.x;
  const int b    = blockIdx.x >> 6;
  const int tile = blockIdx.x & 63;
  const int th0  = (tile >> 3) << 4;
  const int tw0  = (tile & 7) << 4;

  for (int i = tid; i < 18 * 18; i += 256) {
    int yy = i / 18, xx = i % 18;
    int gh = (th0 + yy - 1) & (HDIM - 1);
    int gw = (tw0 + xx - 1) & (WDIM - 1);
    aS[yy][xx] = aws[(b * HDIM + gh) * WDIM + gw];
  }
  __syncthreads();

  const int py = tid >> 4, px = tid & 15;
  const int gh = th0 + py, gw = tw0 + px;
  float amax = -3.4e38f, asum = 0.f;
#pragma unroll
  for (int j = 0; j < 3; ++j)
#pragma unroll
    for (int i = 0; i < 3; ++i) {
      float v = aS[py + j][px + i];
      amax = fmaxf(amax, v); asum += v;
    }
  int post = (amax > 0.1f) && (asum * (1.f / 9.f) < 0.2f);
  int live = post && pre[(b * HDIM + gh) * WDIM + gw];
  if (!live) {
    // dead pixel: zero all 16 channels (live pixels already hold final x_new)
#pragma unroll
    for (int c = 0; c < CCH; ++c)
      out[((size_t)(b * CCH + c) * HDIM + gh) * WDIM + gw] = 0.f;
  }
}

extern "C" void kernel_launch(void* const* d_in, const int* in_sizes, int n_in,
                              void* d_out, int out_size, void* d_ws, size_t ws_size,
                              hipStream_t stream) {
  const float* x       = (const float*)d_in[0];
  const float* rmask   = (const float*)d_in[1];
  const float* filters = (const float*)d_in[2];
  const float* W1      = (const float*)d_in[3];
  const float* b1      = (const float*)d_in[4];
  const float* W2      = (const float*)d_in[5];
  const float* b2      = (const float*)d_in[6];

  float* aws = (float*)d_ws;                                   // 2 MB alpha plane
  unsigned char* pre = (unsigned char*)d_ws +
                       (size_t)32 * HDIM * WDIM * sizeof(float);

  dim3 grid(32 * 64), block(256);
  nca_step_kernel<<<grid, block, 0, stream>>>(x, rmask, filters, W1, b1, W2, b2,
                                              (float*)d_out, aws, pre);
  nca_final_kernel<<<grid, block, 0, stream>>>(aws, pre, (float*)d_out);
}
